// mutiFreq_34050500723359
// MI455X (gfx1250) — matrix operand order repaired and checked by reading, not compiled, not run on hardware
//
#include <hip/hip_runtime.h>

typedef unsigned short u16;
typedef __attribute__((ext_vector_type(16))) __bf16 v16bf;
typedef __attribute__((ext_vector_type(8)))  float  v8f;

// ---------- bf16 helpers (round-to-nearest-even) ----------
__device__ __forceinline__ u16 f2bfbits(float f) {
    unsigned u = __builtin_bit_cast(unsigned, f);
    u += 0x7fffu + ((u >> 16) & 1u);
    return (u16)(u >> 16);
}
__device__ __forceinline__ __bf16 bits2bf(u16 s) {
    return __builtin_bit_cast(__bf16, s);
}
__device__ __forceinline__ __bf16 f2bf(float f) { return bits2bf(f2bfbits(f)); }

// ---------- CDNA5 async global->LDS copy (ASYNCcnt path) ----------
// 16 bytes per lane per issue; LDS offset = low 32 bits of generic LDS ptr.
__device__ __forceinline__ void async_b128(const void* gsrc, void* ldst) {
    unsigned lds = (unsigned)(uintptr_t)ldst;
    unsigned long long g = (unsigned long long)(uintptr_t)gsrc;
    asm volatile("global_load_async_to_lds_b128 %0, %1, off"
                 :: "v"(lds), "v"(g) : "memory");
}
__device__ __forceinline__ void wait_async0() {
    asm volatile("s_wait_asynccnt 0x0" ::: "memory");
}

// ---------- sizes ----------
constexpr int B  = 32;
constexpr int C  = 64;
constexpr int N  = 4096;   // 64*64
constexpr int M  = 1024;   // 32*32
constexpr int TQ = 32;     // q-tile rows per workgroup

// ======================================================================
// Kernel 1: depthwise 3x3 qkv_h : high_f (b,64,64,64) -> q_bf16, vh_bf16
// ======================================================================
__global__ void k_qkv_h(const float* __restrict__ x, const float* __restrict__ w,
                        u16* __restrict__ qo, u16* __restrict__ vo) {
    int idx = blockIdx.x * 256 + threadIdx.x;        // 32*128*4096 total
    int n  = idx & 4095;
    int o  = (idx >> 12) & 127;
    int bi = idx >> 19;
    int y = n >> 6, xx = n & 63;
    int ic = o >> 1;
    const float* xp = x + (((bi << 6) + ic) << 12);
    const float* wp = w + o * 9;
    float acc = 0.f;
#pragma unroll
    for (int ky = 0; ky < 3; ++ky) {
        int yy = y + ky - 1;
        if (yy < 0 || yy > 63) continue;
#pragma unroll
        for (int kx = 0; kx < 3; ++kx) {
            int xc = xx + kx - 1;
            if (xc < 0 || xc > 63) continue;
            acc += xp[(yy << 6) + xc] * wp[ky * 3 + kx];
        }
    }
    u16 hb = f2bfbits(acc);
    if (o < 64) qo[(((bi << 6) + o) << 12) + n] = hb;
    else        vo[(((bi << 6) + (o - 64)) << 12) + n] = hb;
}

// ======================================================================
// Kernel 2: 1x1 qkv_l : low_f (b,64,32,32) -> k_bf16, vl_bf16 (+bias)
// ======================================================================
__global__ void k_qkv_l(const float* __restrict__ lf, const float* __restrict__ w,
                        const float* __restrict__ bias,
                        u16* __restrict__ ko, u16* __restrict__ vo) {
    int idx = blockIdx.x * 256 + threadIdx.x;        // 32*128*1024 total
    int m  = idx & 1023;
    int o  = (idx >> 10) & 127;
    int bi = idx >> 17;
    const float* xp = lf + ((bi << 6) << 10);
    const float* wp = w + (o << 6);
    float acc = bias[o];
#pragma unroll 8
    for (int ic = 0; ic < 64; ++ic) acc += xp[(ic << 10) + m] * wp[ic];
    u16 hb = f2bfbits(acc);
    if (o < 64) ko[(((bi << 6) + o) << 10) + m] = hb;
    else        vo[(((bi << 6) + (o - 64)) << 10) + m] = hb;
}

// ======================================================================
// Kernel 3: fused flash attention (per batch, per 32-row q-tile)
//   S = Q^T K (WMMA bf16) -> row softmax * c^-1/2 -> two WMMA GEMMs.
//   The 128KB K-panel LDS region is reused for the vl panel after stage 2;
//   that copy is issued async and overlaps the softmax pass.
// LDS: K/VL panel 128KB + S panel 128KB + Q tile 4KB = 260KB (dynamic)
// ======================================================================
__global__ __launch_bounds__(256)
void k_attn(const u16* __restrict__ qg, const u16* __restrict__ vhg,
            const u16* __restrict__ kg, const u16* __restrict__ vlg,
            float* __restrict__ high_attn, float* __restrict__ lowacc) {
    extern __shared__ char smem[];
    u16*   ldsK = (u16*)smem;                            // [64][1024] bf16 (K, then VL)
    float* S    = (float*)(smem + C * M * 2);            // [32][1024] f32
    u16*   ldsQ = (u16*)(smem + C * M * 2 + TQ * M * 4); // [32][64] bf16 (q-major)

    const int qt  = blockIdx.x;
    const int bi  = blockIdx.y;
    const int q0  = qt * TQ;
    const int tid = threadIdx.x;
    const int lane = tid & 31;
    const int wv   = tid >> 5;                           // 8 waves

    const u16* qb  = qg  + bi * C * N;
    const u16* vhb = vhg + bi * C * N;
    const u16* kb  = kg  + bi * C * M;
    const u16* vlb = vlg + bi * C * M;

    // ---- async-stage K panel into LDS (b128 per lane) ----
    {
        const uint4* src = (const uint4*)kb;
        uint4* dst = (uint4*)ldsK;
        for (int i = tid; i < (C * M) / 8; i += 256) async_b128(src + i, dst + i);
    }
    // ---- stage Q tile transposed: ldsQ[r][cc] = q[cc][q0+r] ----
    for (int e = tid; e < TQ * C; e += 256) {
        int r = e >> 6, cc = e & 63;
        ldsQ[e] = qb[cc * N + q0 + r];
    }
    wait_async0();
    __syncthreads();

    // ---- Stage 2: S[q,k] = sum_c Q[c,q] K[c,k]  (2 qsub x 64 ksub tiles) ----
    for (int t = wv; t < 128; t += 8) {
        int qs = t >> 6, ks = t & 63;
        v8f c = {};
#pragma unroll
        for (int c0 = 0; c0 < 64; c0 += 32) {
            v16bf a, b;
            int l = lane & 15, kbs = (lane >> 4) << 3;   // ISA A-layout K split
            const u16* arow = ldsQ + (qs * 16 + l) * 64 + c0;
#pragma unroll
            for (int i = 0; i < 8; ++i) {
                a[i]     = bits2bf(arow[kbs + i]);
                a[8 + i] = bits2bf(arow[16 + kbs + i]);
            }
            const u16* brow = ldsK + c0 * M + ks * 16 + l; // one column per lane, K in table order
#pragma unroll
            for (int i = 0; i < 16; ++i) b[i] = bits2bf(brow[(kbs + (i & 7) + ((i >> 3) << 4)) * M]);
            c = __builtin_amdgcn_wmma_f32_16x16x32_bf16(false, a, false, b,
                                                        (short)0, c, false, false);
        }
        int m0 = (lane >> 4) << 3, nn = lane & 15;
#pragma unroll
        for (int i = 0; i < 8; ++i)
            S[(qs * 16 + m0 + i) * M + ks * 16 + nn] = c[i];
    }
    __syncthreads();

    // ---- overlap: async-stage VL panel into the (now dead) K region ----
    {
        const uint4* src = (const uint4*)vlb;
        uint4* dst = (uint4*)ldsK;
        for (int i = tid; i < (C * M) / 8; i += 256) async_b128(src + i, dst + i);
    }

    // ---- Stage 3: row softmax over M, then * c^-1/2 (=0.125) ----
    {
        int row = tid >> 3, sub = tid & 7;               // 8 lanes per row
        float* sr = S + row * M;
        float mx = -3.4e38f;
        for (int j = sub; j < M; j += 8) mx = fmaxf(mx, sr[j]);
        mx = fmaxf(mx, __shfl_xor(mx, 1, 8));
        mx = fmaxf(mx, __shfl_xor(mx, 2, 8));
        mx = fmaxf(mx, __shfl_xor(mx, 4, 8));
        float sum = 0.f;
        for (int j = sub; j < M; j += 8) { float e = __expf(sr[j] - mx); sr[j] = e; sum += e; }
        sum += __shfl_xor(sum, 1, 8);
        sum += __shfl_xor(sum, 2, 8);
        sum += __shfl_xor(sum, 4, 8);
        float inv = 0.125f / sum;
        for (int j = sub; j < M; j += 8) sr[j] *= inv;
    }
    wait_async0();
    __syncthreads();

    // ---- Stage 4: high path  D[q, cc] = sum_k P[q,k] * vl[cc,k]  (VL from LDS) ----
    {
        int qs = wv >> 2, cs = wv & 3;                   // 2 x 4 tiles, 1 per wave
        v8f c = {};
        for (int kk0 = 0; kk0 < M; kk0 += 32) {
            v16bf a, b;
            int l = lane & 15, kbs = (lane >> 4) << 3;
            const float* arow = S + (qs * 16 + l) * M + kk0;
#pragma unroll
            for (int i = 0; i < 8; ++i) {
                a[i]     = f2bf(arow[kbs + i]);
                a[8 + i] = f2bf(arow[16 + kbs + i]);
            }
#pragma unroll
            for (int i = 0; i < 16; ++i)                 // B[k][n=l] = vl[cs*16+l][kk0+k], k in table order
                b[i] = bits2bf(ldsK[(cs * 16 + l) * M + kk0 + kbs + (i & 7) + ((i >> 3) << 4)]);
            c = __builtin_amdgcn_wmma_f32_16x16x32_bf16(false, a, false, b,
                                                        (short)0, c, false, false);
        }
        int m0 = (lane >> 4) << 3, nn = lane & 15;
#pragma unroll
        for (int i = 0; i < 8; ++i)
            high_attn[(bi * C + cs * 16 + nn) * N + q0 + qs * 16 + m0 + i] = c[i];
    }

    // ---- Stage 5: low path  D[cc, kk] += sum_q vh[cc,q] * P[q,kk]  (K=32) ----
    {
        int cs = wv & 3;
        v16bf a;
        int l = lane & 15, kbs = (lane >> 4) << 3;
        const u16* arow = vhb + (cs * 16 + l) * N + q0;  // A[r,kq] = vh[cs*16+r][q0+kq]
#pragma unroll
        for (int i = 0; i < 8; ++i) {
            a[i]     = bits2bf(arow[kbs + i]);
            a[8 + i] = bits2bf(arow[16 + kbs + i]);
        }
        for (int ks = (wv >> 2); ks < 64; ks += 2) {
            v16bf b;
#pragma unroll
            for (int i = 0; i < 16; ++i)                 // B[k][n=l] = P[k][ks*16+l], k in table order
                b[i] = f2bf(S[(kbs + (i & 7) + ((i >> 3) << 4)) * M + ks * 16 + l]);
            v8f c = {};
            c = __builtin_amdgcn_wmma_f32_16x16x32_bf16(false, a, false, b,
                                                        (short)0, c, false, false);
            int m0 = (lane >> 4) << 3, nn = lane & 15;
#pragma unroll
            for (int i = 0; i < 8; ++i)
                unsafeAtomicAdd(&lowacc[(bi * C + cs * 16 + m0 + i) * M + ks * 16 + nn], c[i]);
        }
    }
}

// ======================================================================
// Kernel 4: depthwise 3x3 out_h : high_attn f32 -> d_out (high)
// ======================================================================
__global__ void k_out_h(const float* __restrict__ x, const float* __restrict__ w,
                        float* __restrict__ out) {
    int idx = blockIdx.x * 256 + threadIdx.x;        // 32*64*4096 total
    int n  = idx & 4095;
    int o  = (idx >> 12) & 63;
    int bi = idx >> 18;
    int y = n >> 6, xx = n & 63;
    const float* xp = x + (((bi << 6) + o) << 12);
    const float* wp = w + o * 9;
    float acc = 0.f;
#pragma unroll
    for (int ky = 0; ky < 3; ++ky) {
        int yy = y + ky - 1;
        if (yy < 0 || yy > 63) continue;
#pragma unroll
        for (int kx = 0; kx < 3; ++kx) {
            int xc = xx + kx - 1;
            if (xc < 0 || xc > 63) continue;
            acc += xp[(yy << 6) + xc] * wp[ky * 3 + kx];
        }
    }
    out[idx] = acc;
}

// ======================================================================
// Kernel 5: 1x1 out_l : lowacc f32 -> d_out (low) (+bias)
// ======================================================================
__global__ void k_out_l(const float* __restrict__ x, const float* __restrict__ w,
                        const float* __restrict__ bias, float* __restrict__ out) {
    int idx = blockIdx.x * 256 + threadIdx.x;        // 32*64*1024 total
    int m  = idx & 1023;
    int o  = (idx >> 10) & 63;
    int bi = idx >> 16;
    const float* xp = x + ((bi << 6) << 10);
    const float* wp = w + (o << 6);
    float acc = bias[o];
#pragma unroll 8
    for (int ic = 0; ic < 64; ++ic) acc += xp[(ic << 10) + m] * wp[ic];
    out[idx] = acc;
}

// ======================================================================
extern "C" void kernel_launch(void* const* d_in, const int* in_sizes, int n_in,
                              void* d_out, int out_size, void* d_ws, size_t ws_size,
                              hipStream_t stream) {
    (void)in_sizes; (void)n_in; (void)out_size; (void)ws_size;
    const float* high_f  = (const float*)d_in[0];
    const float* low_f   = (const float*)d_in[1];
    const float* w_qkv_h = (const float*)d_in[2];
    const float* w_qkv_l = (const float*)d_in[3];
    const float* b_qkv_l = (const float*)d_in[4];
    const float* w_out_h = (const float*)d_in[5];
    const float* w_out_l = (const float*)d_in[6];
    const float* b_out_l = (const float*)d_in[7];

    float* out_high = (float*)d_out;                 // b*c*N
    float* out_low  = out_high + B * C * N;          // b*c*M

    // workspace layout (~80 MB total)
    char* ws = (char*)d_ws;
    u16*   q_bf      = (u16*)ws;                                 // 16 MB
    u16*   vh_bf     = q_bf  + (size_t)B * C * N;                // 16 MB
    u16*   k_bf      = vh_bf + (size_t)B * C * N;                //  4 MB
    u16*   vl_bf     = k_bf  + (size_t)B * C * M;                //  4 MB
    float* high_attn = (float*)(vl_bf + (size_t)B * C * M);      // 32 MB
    float* lowacc    = high_attn + (size_t)B * C * N;            //  8 MB

    hipMemsetAsync(lowacc, 0, (size_t)B * C * M * sizeof(float), stream);

    k_qkv_h<<<(B * 2 * C * N) / 256, 256, 0, stream>>>(high_f, w_qkv_h, q_bf, vh_bf);
    k_qkv_l<<<(B * 2 * C * M) / 256, 256, 0, stream>>>(low_f, w_qkv_l, b_qkv_l, k_bf, vl_bf);

    dim3 g(N / TQ, B);                               // 128 x 32 workgroups
    size_t lds = (size_t)C * M * 2 + (size_t)TQ * M * 4 + (size_t)TQ * C * 2; // 266240 B
    k_attn<<<g, 256, lds, stream>>>(q_bf, vh_bf, k_bf, vl_bf, high_attn, lowacc);

    k_out_h<<<(B * C * N) / 256, 256, 0, stream>>>(high_attn, w_out_h, out_high);
    k_out_l<<<(B * C * M) / 256, 256, 0, stream>>>(lowacc, w_out_l, b_out_l, out_low);
}
